// Block_16887811408352
// MI455X (gfx1250) — matrix operand order repaired and checked by reading, not compiled, not run on hardware
//
#include <hip/hip_runtime.h>
#include <hip/hip_bf16.h>
#include <stdint.h>

// ---------------------------------------------------------------------------
// Transformer block for MI455X (gfx1250, wave32, WMMA 16x16x32 f16->f32)
//   B=2, T=2048, D=1024, H=16, DH=64
// GEMMs: async double-buffered LDS staging (GLOBAL_LOAD_ASYNC_TO_LDS_B128,
// ASYNCcnt) + 8 v_wmma per wave per stage. Attention: flash-style online
// softmax with WMMA QK^T and P*V.
// ---------------------------------------------------------------------------

typedef __attribute__((ext_vector_type(16))) _Float16 v16h;
typedef __attribute__((ext_vector_type(8)))  float    v8f;

union FragU {
    uint32_t u[8];
    v16h     h;
};

static __device__ __forceinline__ v8f wmma_f16(const FragU& a, const FragU& b, v8f c) {
    return __builtin_amdgcn_wmma_f32_16x16x32_f16(
        /*neg_a=*/false, a.h, /*neg_b=*/false, b.h,
        /*c_mod=*/(short)0, c, /*reuse_a=*/false, /*reuse_b=*/false);
}

// Async DMA: 16 bytes global -> LDS, tracked by ASYNCcnt.
// %0 = VGPR holding LDS byte address, %1 = VGPR pair with 64-bit global addr.
static __device__ __forceinline__ void async_ld_b128(uint32_t lds_addr, const void* gaddr) {
    asm volatile("global_load_async_to_lds_b128 %0, %1, off"
                 :: "v"(lds_addr), "v"((uint64_t)(size_t)gaddr)
                 : "memory");
}
static __device__ __forceinline__ void wait_async_le6() {
    asm volatile("s_wait_asynccnt 0x6" ::: "memory");
}
static __device__ __forceinline__ void wait_async_0() {
    asm volatile("s_wait_asynccnt 0x0" ::: "memory");
}

#define Bdim 2
#define Tdim 2048
#define Ddim 1024
#define Hdim 16
#define DHdim 64

// ---------------------------------------------------------------------------
// Weight prep: wq/wk/wv f32 [H,D,DH] -> wqkvT f16 [3072][1024]  (row n, col d)
// ---------------------------------------------------------------------------
__global__ __launch_bounds__(256) void prep_qkvw_kernel(
    const float* __restrict__ wq, const float* __restrict__ wk,
    const float* __restrict__ wv, _Float16* __restrict__ wt)
{
    int idx = blockIdx.x * 256 + threadIdx.x;       // 0 .. 3072*1024-1
    int n = idx >> 10;                              // output row
    int d = idx & 1023;                             // output col (K dim)
    int r = n >> 10;                                // 0=q 1=k 2=v
    int nn = n & 1023;
    int h = nn >> 6, k = nn & 63;
    const float* w = (r == 0) ? wq : (r == 1) ? wk : wv;
    wt[idx] = (_Float16)w[((size_t)h * Ddim + d) * DHdim + k];
}

// Generic transpose+convert: in f32 [K,N] -> out f16 [N,K]
__global__ __launch_bounds__(256) void transpose_f32_f16_kernel(
    const float* __restrict__ in, _Float16* __restrict__ out, int K, int N)
{
    int idx = blockIdx.x * 256 + threadIdx.x;       // over N*K
    int n = idx / K;
    int k = idx % K;
    out[idx] = (_Float16)in[(size_t)k * N + n];
}

// ---------------------------------------------------------------------------
// LayerNorm: one row (1024) per 256-thread block, output f16
// ---------------------------------------------------------------------------
__global__ __launch_bounds__(256) void ln_kernel(
    const float* __restrict__ x, const float* __restrict__ g,
    const float* __restrict__ beta, _Float16* __restrict__ out)
{
    __shared__ float red[8];
    int row = blockIdx.x;
    const float* xr = x + (size_t)row * Ddim;
    int tid = threadIdx.x;
    float v[4];
    float s = 0.f;
#pragma unroll
    for (int j = 0; j < 4; ++j) { v[j] = xr[tid + j * 256]; s += v[j]; }
#pragma unroll
    for (int off = 16; off >= 1; off >>= 1) s += __shfl_xor(s, off, 32);
    int w = tid >> 5, lane = tid & 31;
    if (lane == 0) red[w] = s;
    __syncthreads();
    float mean = 0.f;
#pragma unroll
    for (int i = 0; i < 8; ++i) mean += red[i];
    mean *= (1.0f / (float)Ddim);
    __syncthreads();
    float s2 = 0.f;
#pragma unroll
    for (int j = 0; j < 4; ++j) { float d = v[j] - mean; s2 += d * d; }
#pragma unroll
    for (int off = 16; off >= 1; off >>= 1) s2 += __shfl_xor(s2, off, 32);
    if (lane == 0) red[w] = s2;
    __syncthreads();
    float var = 0.f;
#pragma unroll
    for (int i = 0; i < 8; ++i) var += red[i];
    var *= (1.0f / (float)Ddim);
    float rstd = rsqrtf(var + 1e-5f);
#pragma unroll
    for (int j = 0; j < 4; ++j) {
        int c = tid + j * 256;
        out[(size_t)row * Ddim + c] = (_Float16)((v[j] - mean) * rstd * g[c] + beta[c]);
    }
}

// ---------------------------------------------------------------------------
// WMMA GEMM:  C[M,N] = A[M,K] * Bt[N,K]^T (+bias)(+residual)(relu?)
// A,Bt f16 row-major, 16B-aligned rows. 256 threads = 8 waves (2M x 4N),
// block tile 64x128, wave tile 32x32, K chunk 64.
// Async double-buffered LDS staging; 8 WMMAs per wave per stage.
// ---------------------------------------------------------------------------
#define KC   64
#define LSTR 36   // LDS row stride in u32: 32 data + 4 pad (keeps 16B align)

__global__ __launch_bounds__(256) void gemm_wmma_kernel(
    const _Float16* __restrict__ A, const _Float16* __restrict__ Bt,
    const float* __restrict__ bias, const float* __restrict__ residual,
    float* __restrict__ outf, _Float16* __restrict__ outh,
    int N, int Kd, int relu)
{
    __shared__ uint32_t As[2][64 * LSTR];    // 18.0 KB
    __shared__ uint32_t Bs[2][128 * LSTR];   // 36.0 KB

    int tid  = threadIdx.x;
    int lane = tid & 31;
    int w    = tid >> 5;               // wave 0..7
    int wm   = w & 1;                  // M sub-tile (x32)
    int wn   = w >> 1;                 // N sub-tile (x32)
    int bm   = blockIdx.x * 64;
    int bn   = blockIdx.y * 128;

    int  m16 = lane & 15;
    bool hih = lane >= 16;

    uint32_t asb = (uint32_t)(uintptr_t)&As[0][0];   // LDS byte offsets
    uint32_t bsb = (uint32_t)(uintptr_t)&Bs[0][0];

    // Per-thread async staging of one K-chunk (64 halves = 8 x 16B per row):
    //   A: 64 rows * 8 chunks = 512 -> 2 per thread
    //   B: 128 rows * 8 chunks = 1024 -> 4 per thread   (6 async ops total)
    auto stage = [&](int kc, int buf) {
        uint32_t ab = asb + (uint32_t)buf * (64 * LSTR * 4);
        uint32_t bb = bsb + (uint32_t)buf * (128 * LSTR * 4);
#pragma unroll
        for (int j = 0; j < 2; ++j) {
            int id = tid + j * 256;
            int r = id >> 3, c = id & 7;
            async_ld_b128(ab + (uint32_t)(r * LSTR + c * 4) * 4,
                          A + (size_t)(bm + r) * Kd + kc + c * 8);
        }
#pragma unroll
        for (int j = 0; j < 4; ++j) {
            int id = tid + j * 256;
            int r = id >> 3, c = id & 7;
            async_ld_b128(bb + (uint32_t)(r * LSTR + c * 4) * 4,
                          Bt + (size_t)(bn + r) * Kd + kc + c * 8);
        }
    };

    v8f c00 = {}, c01 = {}, c10 = {}, c11 = {};

    int nk = Kd / KC;
    stage(0, 0);
    for (int t = 0; t < nk; ++t) {
        int buf = t & 1;
        if (t + 1 < nk) { stage((t + 1) * KC, buf ^ 1); wait_async_le6(); }
        else           { wait_async_0(); }
        __syncthreads();

        const uint32_t* Ab = &As[buf][0];
        const uint32_t* Bb = &Bs[buf][0];
#pragma unroll
        for (int ks = 0; ks < 2; ++ks) {
            FragU a0, a1, b0, b1;
#pragma unroll
            for (int i = 0; i < 8; ++i) {
                int k0 = ((i < 4) ? 2 * i : 16 + 2 * (i - 4)) + (hih ? 8 : 0);
                int cu = (ks * 32 + k0) >> 1;
                a0.u[i] = Ab[(wm * 32 + m16) * LSTR + cu];
                a1.u[i] = Ab[(wm * 32 + 16 + m16) * LSTR + cu];
                int kk = cu;
                b0.u[i] = Bb[(wn * 32 + m16) * LSTR + kk];
                b1.u[i] = Bb[(wn * 32 + 16 + m16) * LSTR + kk];
            }
            c00 = wmma_f16(a0, b0, c00);
            c01 = wmma_f16(a0, b1, c01);
            c10 = wmma_f16(a1, b0, c10);
            c11 = wmma_f16(a1, b1, c11);
        }
        __syncthreads();
    }

    int mrow = hih ? 8 : 0;
#pragma unroll
    for (int sm = 0; sm < 2; ++sm)
#pragma unroll
        for (int sn = 0; sn < 2; ++sn) {
            v8f acc = (sm == 0) ? (sn == 0 ? c00 : c01) : (sn == 0 ? c10 : c11);
            int col = bn + wn * 32 + sn * 16 + m16;
            float bv = bias ? bias[col] : 0.f;
#pragma unroll
            for (int i = 0; i < 8; ++i) {
                int row = bm + wm * 32 + sm * 16 + i + mrow;
                float v = acc[i] + bv;
                if (residual) v += residual[(size_t)row * N + col];
                if (relu) v = fmaxf(v, 0.f);
                if (outf) outf[(size_t)row * N + col] = v;
                if (outh) outh[(size_t)row * N + col] = (_Float16)v;
            }
        }
}

// ---------------------------------------------------------------------------
// Vt transpose: qkv f16 [B*T, 3072] (V at col 2048+) -> vt f16 [B,H,DH,T]
// ---------------------------------------------------------------------------
__global__ __launch_bounds__(256) void vt_kernel(
    const _Float16* __restrict__ qkv, _Float16* __restrict__ vt)
{
    int idx  = blockIdx.x * 256 + threadIdx.x;     // B*H*DH*T = 4194304
    int s    = idx & (Tdim - 1);
    int rest = idx >> 11;
    int k    = rest & 63;
    int bh   = rest >> 6;
    int h    = bh & (Hdim - 1);
    int b    = bh >> 4;
    vt[idx] = qkv[((size_t)(b * Tdim + s)) * 3072 + 2048 + h * DHdim + k];
}

// ---------------------------------------------------------------------------
// Flash attention: one wave per (b,h, 16-row tile). s-tiles of 32 columns.
// QK^T via 2 WMMAs per N-subtile (K=DH=64); P(16x32) -> LDS -> A-frag;
// P*V via 4 WMMAs (DH = 4 x 16).
// ---------------------------------------------------------------------------
__global__ __launch_bounds__(32) void attn_kernel(
    const _Float16* __restrict__ qkv, const _Float16* __restrict__ vt,
    _Float16* __restrict__ attout)
{
    __shared__ _Float16 p_lds[16 * 34];            // 16 rows x 32 (+2 pad)

    const int NT = Tdim / 16;                      // 128
    int tile = blockIdx.x % NT;
    int bh   = blockIdx.x / NT;
    int h    = bh & (Hdim - 1);
    int b    = bh >> 4;
    int t0   = tile * 16;

    int  lane = threadIdx.x;
    int  m16  = lane & 15;
    bool hih  = lane >= 16;
    int  mrow = hih ? 8 : 0;

    // Q fragments (16 rows x 64 K, two K-chunks of 32)
    FragU qa[2];
    const _Float16* qbase = qkv + ((size_t)(b * Tdim + t0)) * 3072 + h * DHdim;
#pragma unroll
    for (int c = 0; c < 2; ++c)
#pragma unroll
        for (int i = 0; i < 8; ++i) {
            int k0 = ((i < 4) ? 2 * i : 16 + 2 * (i - 4)) + (hih ? 8 : 0);
            qa[c].u[i] = *(const uint32_t*)(qbase + (size_t)m16 * 3072 + c * 32 + k0);
        }

    v8f o[4] = {{}, {}, {}, {}};
    float mprev[8], lsum[8];
#pragma unroll
    for (int i = 0; i < 8; ++i) { mprev[i] = -__builtin_inff(); lsum[i] = 0.f; }
    const float scale = 0.125f;                    // DH^-0.5

    for (int s0 = 0; s0 <= t0; s0 += 32) {
        // S = Q K^T over a 16x32 block of columns
        v8f sa = {}, sb = {};
        const _Float16* kbp = qkv + ((size_t)(b * Tdim + s0)) * 3072 + 1024 + h * DHdim;
#pragma unroll
        for (int c = 0; c < 2; ++c) {
            FragU kb0, kb1;
#pragma unroll
            for (int i = 0; i < 8; ++i) {
                int kk = ((i < 4) ? 2 * i : 16 + 2 * (i - 4)) + (hih ? 8 : 0);
                kb0.u[i] = *(const uint32_t*)(kbp + (size_t)m16 * 3072 + c * 32 + kk);
                kb1.u[i] = *(const uint32_t*)(kbp + (size_t)(m16 + 16) * 3072 + c * 32 + kk);
            }
            sa = wmma_f16(qa[c], kb0, sa);
            sb = wmma_f16(qa[c], kb1, sb);
        }

        bool need_mask = (s0 + 31 > t0);
        float p0[8], p1[8], alpha[8];
#pragma unroll
        for (int i = 0; i < 8; ++i) {
            float v0 = sa[i] * scale, v1 = sb[i] * scale;
            if (need_mask) {
                int row = t0 + i + mrow;
                if (s0 + m16 > row)      v0 = -__builtin_inff();
                if (s0 + 16 + m16 > row) v1 = -__builtin_inff();
            }
            float mx = fmaxf(v0, v1);
#pragma unroll
            for (int off = 1; off < 16; off <<= 1) mx = fmaxf(mx, __shfl_xor(mx, off, 32));
            float mnew = fmaxf(mprev[i], mx);
            float a  = __expf(mprev[i] - mnew);
            float e0 = __expf(v0 - mnew);
            float e1 = __expf(v1 - mnew);
            float rs = e0 + e1;
#pragma unroll
            for (int off = 1; off < 16; off <<= 1) rs += __shfl_xor(rs, off, 32);
            lsum[i]  = lsum[i] * a + rs;
            mprev[i] = mnew;
            p0[i] = e0; p1[i] = e1; alpha[i] = a;
        }
#pragma unroll
        for (int j = 0; j < 4; ++j)
#pragma unroll
            for (int i = 0; i < 8; ++i) o[j][i] *= alpha[i];

        // P (C-layout) -> LDS -> A-fragment layout
        __syncthreads();
#pragma unroll
        for (int i = 0; i < 8; ++i) {
            int r = i + mrow;
            p_lds[r * 34 + m16]      = (_Float16)p0[i];
            p_lds[r * 34 + 16 + m16] = (_Float16)p1[i];
        }
        __syncthreads();
        FragU pa;
#pragma unroll
        for (int i = 0; i < 8; ++i) {
            int k0 = ((i < 4) ? 2 * i : 16 + 2 * (i - 4)) + (hih ? 8 : 0);
            pa.u[i] = *(const uint32_t*)&p_lds[m16 * 34 + k0];
        }

        // O += P * V  (V as Vt[B,H,DH,T]: rows = output cols, contiguous in s)
        const _Float16* vbp = vt + ((size_t)bh * DHdim) * Tdim + s0;
#pragma unroll
        for (int j = 0; j < 4; ++j) {
            FragU vb;
#pragma unroll
            for (int i = 0; i < 8; ++i) {
                int kk = ((i < 4) ? 2 * i : 16 + 2 * (i - 4)) + (hih ? 8 : 0);
                vb.u[i] = *(const uint32_t*)(vbp + (size_t)(j * 16 + m16) * Tdim + kk);
            }
            o[j] = wmma_f16(pa, vb, o[j]);
        }
    }

    float rinv[8];
#pragma unroll
    for (int i = 0; i < 8; ++i) rinv[i] = 1.f / lsum[i];
#pragma unroll
    for (int j = 0; j < 4; ++j)
#pragma unroll
        for (int i = 0; i < 8; ++i) {
            int row = t0 + i + mrow;
            attout[((size_t)(b * Tdim + row)) * Ddim + h * DHdim + j * 16 + m16] =
                (_Float16)(o[j][i] * rinv[i]);
        }
}

// ---------------------------------------------------------------------------
// Launch
// ---------------------------------------------------------------------------
extern "C" void kernel_launch(void* const* d_in, const int* in_sizes, int n_in,
                              void* d_out, int out_size, void* d_ws, size_t ws_size,
                              hipStream_t stream)
{
    const float* x     = (const float*)d_in[0];
    const float* wq    = (const float*)d_in[1];
    const float* wk    = (const float*)d_in[2];
    const float* wv    = (const float*)d_in[3];
    const float* wproj = (const float*)d_in[4];
    const float* bproj = (const float*)d_in[5];
    const float* w1    = (const float*)d_in[6];
    const float* b1    = (const float*)d_in[7];
    const float* w2    = (const float*)d_in[8];
    const float* b2    = (const float*)d_in[9];
    const float* ln1g  = (const float*)d_in[10];
    const float* ln1b  = (const float*)d_in[11];
    const float* ln2g  = (const float*)d_in[12];
    const float* ln2b  = (const float*)d_in[13];

    const int M = Bdim * Tdim;                     // 4096 rows

    char* ws = (char*)d_ws;
    size_t off = 0;
    auto alloc = [&](size_t bytes) -> char* {
        char* p = ws + off;
        off += (bytes + 255) & ~(size_t)255;
        return p;
    };

    _Float16* wqkvT  = (_Float16*)alloc((size_t)3072 * 1024 * 2);
    _Float16* wprojT = (_Float16*)alloc((size_t)1024 * 1024 * 2);
    _Float16* w1T    = (_Float16*)alloc((size_t)4096 * 1024 * 2);
    _Float16* w2T    = (_Float16*)alloc((size_t)1024 * 4096 * 2);
    _Float16* xn1    = (_Float16*)alloc((size_t)M * 1024 * 2);
    _Float16* qkv    = (_Float16*)alloc((size_t)M * 3072 * 2);
    _Float16* vtb    = (_Float16*)alloc((size_t)Bdim * Hdim * DHdim * Tdim * 2);
    _Float16* attout = (_Float16*)alloc((size_t)M * 1024 * 2);
    float*    x1     = (float*)   alloc((size_t)M * 1024 * 4);
    _Float16* xn2    = (_Float16*)alloc((size_t)M * 1024 * 2);
    // FFN hidden aliases the (dead by then) qkv + vt region: 25.2MB + 8.4MB
    _Float16* hbuf   = qkv;

    float* outp = (float*)d_out;

    // weight prep
    prep_qkvw_kernel<<<(3072 * 1024) / 256, 256, 0, stream>>>(wq, wk, wv, wqkvT);
    transpose_f32_f16_kernel<<<(1024 * 1024) / 256, 256, 0, stream>>>(wproj, wprojT, 1024, 1024);
    transpose_f32_f16_kernel<<<(4096 * 1024) / 256, 256, 0, stream>>>(w1, w1T, 1024, 4096);
    transpose_f32_f16_kernel<<<(1024 * 4096) / 256, 256, 0, stream>>>(w2, w2T, 4096, 1024);

    // LN1
    ln_kernel<<<M, 256, 0, stream>>>(x, ln1g, ln1b, xn1);

    // QKV projection: [4096,1024] x [3072,1024]^T -> qkv f16
    gemm_wmma_kernel<<<dim3(M / 64, 3072 / 128), 256, 0, stream>>>(
        xn1, wqkvT, nullptr, nullptr, nullptr, qkv, 3072, 1024, 0);

    // V transpose
    vt_kernel<<<(Bdim * Hdim * DHdim * Tdim) / 256, 256, 0, stream>>>(qkv, vtb);

    // Flash attention
    attn_kernel<<<Bdim * Hdim * (Tdim / 16), 32, 0, stream>>>(qkv, vtb, attout);

    // Out projection + residual 1 -> x1 (f32)
    gemm_wmma_kernel<<<dim3(M / 64, 1024 / 128), 256, 0, stream>>>(
        attout, wprojT, bproj, x, x1, nullptr, 1024, 1024, 0);

    // LN2
    ln_kernel<<<M, 256, 0, stream>>>(x1, ln2g, ln2b, xn2);

    // FFN1: relu(xn2 @ w1 + b1) -> h (f16)
    gemm_wmma_kernel<<<dim3(M / 64, 4096 / 128), 256, 0, stream>>>(
        xn2, w1T, b1, nullptr, nullptr, hbuf, 4096, 1024, 1);

    // FFN2: h @ w2 + b2 + x1 -> out (f32)
    gemm_wmma_kernel<<<dim3(M / 64, 1024 / 128), 256, 0, stream>>>(
        hbuf, w2T, b2, x1, outp, nullptr, 1024, 4096, 0);
}
